// GRUCell_64793876628224
// MI455X (gfx1250) — matrix operand order repaired and checked by reading, compiled, not run on hardware
//
#include <hip/hip_runtime.h>

// ---------------------------------------------------------------------------
// GRU cell, fused, for MI455X (gfx1250, wave32, WMMA).
//
// Roofline: 4.1e11 FLOP vs 288MB unique f32 data -> compute bound. bf16 WMMA
// (16x16x32) has 8x the K-depth of the f32 WMMA (16x16x4). Pass 1 converts
// all operands to bf16 in d_ws (112MB -> resident in the 192MB L2) using
// interleaved layouts chosen so that the GEMM kernel's 28 loads per K-step
// are reachable from TWO per-lane base addresses via 24-bit immediate
// offsets (weight gates 4KB apart, x/h sides 4KB apart, M-subtiles 128KB
// apart -- all < +-8MB). Pass 2 is the fused GEMM+gates kernel.
//
//   xh pack: xh[(row*2 + side)*2048 + k]        side: 0=x, 1=h
//   w  pack: wp[(n*6 + gate)*2048 + k]          gate: ir,hr,iz,hz,in,hn
//
// Register blocking: each wave owns a 64(M)x16(N) tile (four 16x16 M-subtiles
// sharing every B fragment): 24 WMMAs per 448 loaded bytes/lane (~27 FLOP/B
// of VGPR fill), 128 accumulator VGPRs.
// ---------------------------------------------------------------------------

typedef __attribute__((ext_vector_type(16))) __bf16 v16bf;
typedef __attribute__((ext_vector_type(8)))  float  v8f;

#define GRU_BATCH 8192
#define GRU_HID   2048
#define GRU_K     2048

union Frag {
    uint4 q[2];   // two 16B global loads
    v16bf bf;     // WMMA operand view
};

__device__ __forceinline__ float fast_sigmoid(float x) {
    return 1.0f / (1.0f + __expf(-x));
}
__device__ __forceinline__ float fast_tanh(float x) {
    float e = __expf(-2.0f * x);
    return (1.0f - e) / (1.0f + e);
}

// f32 -> bf16 (round to nearest even) with strided row packing:
// dst elem index = dst_base + row*row_stride + col   (rows are 2048 wide).
__global__ __launch_bounds__(256) void cvt_pack_bf16(
    const float* __restrict__ s, unsigned short* __restrict__ d,
    int dst_base, int row_stride, int n) {
    int i = (blockIdx.x * 256 + threadIdx.x) * 4;
    if (i >= n) return;
    float4 f = *(const float4*)(s + i);
    ushort4 o;
    unsigned int b;
    b = __float_as_uint(f.x); o.x = (unsigned short)((b + 0x7FFFu + ((b >> 16) & 1u)) >> 16);
    b = __float_as_uint(f.y); o.y = (unsigned short)((b + 0x7FFFu + ((b >> 16) & 1u)) >> 16);
    b = __float_as_uint(f.z); o.z = (unsigned short)((b + 0x7FFFu + ((b >> 16) & 1u)) >> 16);
    b = __float_as_uint(f.w); o.w = (unsigned short)((b + 0x7FFFu + ((b >> 16) & 1u)) >> 16);
    const int row = i >> 11;          // / 2048
    const int col = i & 2047;
    *(ushort4*)(d + dst_base + row * row_stride + col) = o;
}

__device__ __forceinline__ v8f wmma_bf16(const Frag& a, const Frag& b, v8f c) {
    // 8 args: (neg_a, A, neg_b, B, c_mod, C, reuse_a, reuse_b)
    return __builtin_amdgcn_wmma_f32_16x16x32_bf16(
        false, a.bf, false, b.bf, (short)0, c, false, false);
}

// A fragment (16x32 bf16): chunk pair at +0 / +32 bytes.
__device__ __forceinline__ Frag load_a(const char* p) {
    Frag f;
    f.q[0] = *(const uint4*)(p);
    f.q[1] = *(const uint4*)(p + 32);
    return f;
}
// B fragment (32x16 bf16): chunk pair at +0 / +32 bytes.
__device__ __forceinline__ Frag load_b(const char* p) {
    Frag f;
    f.q[0] = *(const uint4*)(p);
    f.q[1] = *(const uint4*)(p + 32);
    return f;
}

// Fused GEMM + gates. Grid: (BATCH/128, HID/64), 256 threads = 8 waves laid
// out 2(M) x 4(N); each wave computes a 64x16 output tile (4 M-subtiles).
__global__ __launch_bounds__(256) void gru_wmma(
    const unsigned short* __restrict__ wsp,  // bf16: xh-pack | w-pack
    const float* __restrict__ b_ir, const float* __restrict__ b_hr,
    const float* __restrict__ b_iz, const float* __restrict__ b_hz,
    const float* __restrict__ b_in, const float* __restrict__ b_hn,
    const float* __restrict__ h_old, float* __restrict__ out) {

    // Byte strides in the packed layouts.
    const int A_ROW  = 2 * GRU_K * 2;        // 8192B  : one M row (x+h)
    const int A_SIDE = GRU_K * 2;            // 4096B  : x -> h
    const int A_SUB  = 16 * A_ROW;           // 131072B: 16-row M-subtile
    const int B_ROW  = 6 * GRU_K * 2;        // 24576B : one N row (6 gates)
    const int B_GATE = GRU_K * 2;            // 4096B  : gate stride
    const size_t XH_BYTES = (size_t)GRU_BATCH * A_ROW;   // 64MB

    const int lane   = threadIdx.x & 31;
    const int wave   = threadIdx.x >> 5;
    const int m_base = blockIdx.x * 128 + (wave & 1) * 64;   // four 16-row subtiles
    const int n_base = blockIdx.y * 64  + (wave >> 1) * 16;

    const int lr = lane & 15;   // row/col within tile
    const int lh = lane >> 4;   // lane half selects K sub-chunk

    // Two per-lane base addresses; everything else is immediate offsets.
    const char* Abase = (const char*)wsp + (size_t)(m_base + lr) * A_ROW + lh * 16;
    const char* Bbase = (const char*)wsp + XH_BYTES
                      + (size_t)(n_base + lr) * B_ROW + lh * 16;

    v8f aR0 = {}, aR1 = {}, aR2 = {}, aR3 = {};
    v8f aZ0 = {}, aZ1 = {}, aZ2 = {}, aZ3 = {};
    v8f aN0 = {}, aN1 = {}, aN2 = {}, aN3 = {};
    v8f aH0 = {}, aH1 = {}, aH2 = {}, aH3 = {};

#pragma unroll 1
    for (int kb = 0; kb < GRU_K * 2; kb += 64) {   // 64B = 32 bf16 of K
        const char* A = Abase + kb;
        const char* B = Bbase + kb;

        // ---- x-side: 3 B fragments shared by 4 M-subtiles -> 12 WMMAs ----
        Frag a0 = load_a(A + 0 * A_SUB);
        Frag a1 = load_a(A + 1 * A_SUB);
        Frag a2 = load_a(A + 2 * A_SUB);
        Frag a3 = load_a(A + 3 * A_SUB);
        Frag br = load_b(B + 0 * B_GATE);   // W_ir
        Frag bz = load_b(B + 2 * B_GATE);   // W_iz
        Frag bn = load_b(B + 4 * B_GATE);   // W_in
        aR0 = wmma_bf16(a0, br, aR0);  aR1 = wmma_bf16(a1, br, aR1);
        aR2 = wmma_bf16(a2, br, aR2);  aR3 = wmma_bf16(a3, br, aR3);
        aZ0 = wmma_bf16(a0, bz, aZ0);  aZ1 = wmma_bf16(a1, bz, aZ1);
        aZ2 = wmma_bf16(a2, bz, aZ2);  aZ3 = wmma_bf16(a3, bz, aZ3);
        aN0 = wmma_bf16(a0, bn, aN0);  aN1 = wmma_bf16(a1, bn, aN1);
        aN2 = wmma_bf16(a2, bn, aN2);  aN3 = wmma_bf16(a3, bn, aN3);

        // ---- h-side ----
        a0 = load_a(A + 0 * A_SUB + A_SIDE);
        a1 = load_a(A + 1 * A_SUB + A_SIDE);
        a2 = load_a(A + 2 * A_SUB + A_SIDE);
        a3 = load_a(A + 3 * A_SUB + A_SIDE);
        br = load_b(B + 1 * B_GATE);        // W_hr
        bz = load_b(B + 3 * B_GATE);        // W_hz
        bn = load_b(B + 5 * B_GATE);        // W_hn
        aR0 = wmma_bf16(a0, br, aR0);  aR1 = wmma_bf16(a1, br, aR1);
        aR2 = wmma_bf16(a2, br, aR2);  aR3 = wmma_bf16(a3, br, aR3);
        aZ0 = wmma_bf16(a0, bz, aZ0);  aZ1 = wmma_bf16(a1, bz, aZ1);
        aZ2 = wmma_bf16(a2, bz, aZ2);  aZ3 = wmma_bf16(a3, bz, aZ3);
        aH0 = wmma_bf16(a0, bn, aH0);  aH1 = wmma_bf16(a1, bn, aH1);
        aH2 = wmma_bf16(a2, bn, aH2);  aH3 = wmma_bf16(a3, bn, aH3);
    }

    // Epilogue. C/D layout: VGPR j -> (M = m_sub + j + lh*8, N = n_base + lr).
    const int n  = n_base + lr;
    const float br = b_ir[n] + b_hr[n];
    const float bz = b_iz[n] + b_hz[n];
    const float bi = b_in[n];
    const float bh = b_hn[n];

    auto emit = [&](const v8f& R, const v8f& Z, const v8f& N, const v8f& H,
                    int m_sub) {
#pragma unroll
        for (int j = 0; j < 8; ++j) {
            const int row = m_sub + j + lh * 8;
            const float r  = fast_sigmoid(R[j] + br);
            const float z  = fast_sigmoid(Z[j] + bz);
            const float nn = fast_tanh(N[j] + bi + r * (H[j] + bh));
            const float ho = h_old[(size_t)row * GRU_HID + n];
            out[(size_t)row * GRU_HID + n] = (1.0f - z) * nn + z * ho;
        }
    };
    emit(aR0, aZ0, aN0, aH0, m_base);
    emit(aR1, aZ1, aN1, aH1, m_base + 16);
    emit(aR2, aZ2, aN2, aH2, m_base + 32);
    emit(aR3, aZ3, aN3, aH3, m_base + 48);
}

extern "C" void kernel_launch(void* const* d_in, const int* in_sizes, int n_in,
                              void* d_out, int out_size, void* d_ws, size_t ws_size,
                              hipStream_t stream) {
    const float* x    = (const float*)d_in[0];
    const float* h    = (const float*)d_in[1];
    const float* W_ir = (const float*)d_in[2];  const float* b_ir = (const float*)d_in[3];
    const float* W_hr = (const float*)d_in[4];  const float* b_hr = (const float*)d_in[5];
    const float* W_iz = (const float*)d_in[6];  const float* b_iz = (const float*)d_in[7];
    const float* W_hz = (const float*)d_in[8];  const float* b_hz = (const float*)d_in[9];
    const float* W_in = (const float*)d_in[10]; const float* b_in = (const float*)d_in[11];
    const float* W_hn = (const float*)d_in[12]; const float* b_hn = (const float*)d_in[13];

    unsigned short* ws = (unsigned short*)d_ws;
    const int XN = GRU_BATCH * GRU_K;           // 16.78M elems per x/h
    const int WN = GRU_HID   * GRU_K;           // 4.19M elems per W
    const int XH_ELEMS = 2 * XN;                // xh pack size

    // Pack: xh[(row*2+side)*2048+k] then wp[(n*6+gate)*2048+k]; 112MB total.
    struct CV { const float* s; int base; int stride; int n; };
    const CV cv[8] = {
        { x,    0,                       2 * GRU_K, XN },  // side 0
        { h,    GRU_K,                   2 * GRU_K, XN },  // side 1
        { W_ir, XH_ELEMS + 0 * GRU_K,    6 * GRU_K, WN },  // gate 0
        { W_hr, XH_ELEMS + 1 * GRU_K,    6 * GRU_K, WN },  // gate 1
        { W_iz, XH_ELEMS + 2 * GRU_K,    6 * GRU_K, WN },  // gate 2
        { W_hz, XH_ELEMS + 3 * GRU_K,    6 * GRU_K, WN },  // gate 3
        { W_in, XH_ELEMS + 4 * GRU_K,    6 * GRU_K, WN },  // gate 4
        { W_hn, XH_ELEMS + 5 * GRU_K,    6 * GRU_K, WN },  // gate 5
    };
    for (int i = 0; i < 8; ++i) {
        const int blocks = cv[i].n / 1024;      // 4 elems/thread, 256 threads
        cvt_pack_bf16<<<blocks, 256, 0, stream>>>(
            cv[i].s, ws, cv[i].base, cv[i].stride, cv[i].n);
    }

    dim3 grid(GRU_BATCH / 128, GRU_HID / 64);   // 64 x 32 workgroups
    gru_wmma<<<grid, 256, 0, stream>>>(ws,
        b_ir, b_hr, b_iz, b_hz, b_in, b_hn, h, (float*)d_out);
}
